// CJEPA_24008867184687
// MI455X (gfx1250) — matrix operand order repaired and checked by reading, not compiled, not run on hardware
//
#include <hip/hip_runtime.h>
#include <hip/hip_bf16.h>

#define B_    32
#define T_    256
#define DOBS  1024
#define D_    256
#define NVAR  16
#define ALPHA_ 0.7f

typedef __attribute__((ext_vector_type(16))) __bf16 v16bf;
typedef __attribute__((ext_vector_type(8)))  __bf16 v8bf;
typedef __attribute__((ext_vector_type(8)))  float  v8f;

// ---- WMMA helpers -------------------------------------------------------

__device__ __forceinline__ v8f wmma_bf16(v16bf a, v16bf b, v8f c) {
  // D = A(16x32 bf16) * B(32x16 bf16) + C(f32)
  return __builtin_amdgcn_wmma_f32_16x16x32_bf16(
      /*neg_a=*/false, a, /*neg_b=*/false, b,
      /*c_mod=*/(short)0, c, /*reuse_a=*/false, /*reuse_b=*/false);
}

// A fragment from a row-major bf16 tile [16][ld] (LDS or global).
// Lane layout: m = lane&15; lanes<16 hold K = ko+{0..15},
// lanes>=16 hold K = ko+{16..31}.
__device__ __forceinline__ v16bf load_a_frag(const __bf16* tile, int ld, int ko, int lane) {
  int m  = lane & 15;
  int kb = (lane >> 4) << 4;
  const __bf16* p = tile + m * ld + ko + kb;
  v8bf lo = *(const v8bf*)(p);
  v8bf hi = *(const v8bf*)(p + 8);
  return __builtin_shufflevector(lo, hi, 0,1,2,3,4,5,6,7,8,9,10,11,12,13,14,15);
}

// B fragment: B[k][c] = mat[c*ldk + k]  (i.e. mat row-major [C][K], computing X @ mat^T).
// Lane layout: c = c0 + (lane&15); 16 contiguous K per lane starting at ko + 16*(lane>>4).
__device__ __forceinline__ v16bf load_b_frag(const __bf16* mat, int ldk, int c0, int ko, int lane) {
  int c  = c0 + (lane & 15);
  int kb = (lane >> 4) << 4;
  return *(const v16bf*)(mat + (size_t)c * ldk + ko + kb);
}

// ---- Kernel 0: fp32 -> bf16 weight conversion ---------------------------

__global__ void cjepa_cvt_bf16(const float* __restrict__ src, __bf16* __restrict__ dst, int n) {
  int i = blockIdx.x * blockDim.x + threadIdx.x;
  if (i < n) dst[i] = (__bf16)src[i];
}

// ---- Kernel 1: Z = tanh(obs @ W_enc^T + b_enc), bf16 out ----------------
// grid = (B*T)/16 workgroups, 512 threads (16 waves, one 16-col tile each).

__global__ __launch_bounds__(512) void cjepa_encode(
    const float* __restrict__ obs, const __bf16* __restrict__ wenc,
    const float* __restrict__ benc, __bf16* __restrict__ zb) {
  __shared__ __align__(16) __bf16 a_lds[16 * DOBS];   // 32 KB
  const int tid = threadIdx.x;
  const int r0  = blockIdx.x * 16;

  // 16 rows of obs are contiguous: flat copy 16384 f32 -> bf16 LDS
  const float4* src = (const float4*)(obs + (size_t)r0 * DOBS);
  for (int i = 0; i < 8; ++i) {
    int fi = tid + i * 512;
    float4 v = src[fi];
    int e = fi * 4;
    a_lds[e+0] = (__bf16)v.x; a_lds[e+1] = (__bf16)v.y;
    a_lds[e+2] = (__bf16)v.z; a_lds[e+3] = (__bf16)v.w;
  }
  __syncthreads();

  const int lane = tid & 31;
  const int c0   = (tid >> 5) * 16;
  const int cl   = c0 + (lane & 15);
  const int mb   = (lane >> 4) << 3;

  v8f acc = {0,0,0,0,0,0,0,0};
  for (int kk = 0; kk < DOBS / 32; ++kk)
    acc = wmma_bf16(load_a_frag(a_lds, DOBS, kk * 32, lane),
                    load_b_frag(wenc, DOBS, c0, kk * 32, lane), acc);

  const float bias = benc[cl];
#pragma unroll
  for (int j = 0; j < 8; ++j) {
    float z = tanhf(acc[j] + bias);
    zb[(size_t)(r0 + mb + j) * D_ + cl] = (__bf16)z;
  }
}

// ---- Kernel 2: keys/values/queries/attn/normalize -> base slots ---------
// grid = (B*T)/16 workgroups, 512 threads.

__global__ __launch_bounds__(512) void cjepa_slots(
    const __bf16* __restrict__ zb, const __bf16* __restrict__ wkey,
    const __bf16* __restrict__ wval, const __bf16* __restrict__ wq,
    const float* __restrict__ bq, float* __restrict__ out) {
  __shared__ __align__(16) __bf16 z_lds[16 * D_];     // 8 KB
  __shared__ float keys[16 * D_];                     // 16 KB
  __shared__ float vals[16 * D_];                     // 16 KB
  __shared__ float q[16 * D_];                        // 16 KB
  __shared__ float red[16 * 32];
  __shared__ float attn_s[16];
  __shared__ float inv_s[16];

  const int tid = threadIdx.x;
  const int r0  = blockIdx.x * 16;

  { // load Z tile (4096 bf16 contiguous)
    const unsigned int* s32 = (const unsigned int*)(zb + (size_t)r0 * D_);
    unsigned int* d32 = (unsigned int*)z_lds;
    for (int i = 0; i < 4; ++i) d32[tid + i * 512] = s32[tid + i * 512];
  }
  __syncthreads();

  const int lane = tid & 31;
  const int c0   = (tid >> 5) * 16;
  const int cl   = c0 + (lane & 15);
  const int mb   = (lane >> 4) << 3;
  const int row  = tid >> 5;   // reduction-phase row
  const int seg  = tid & 31;   // reduction-phase segment (8 cols each)

  { // keys = z @ W_key^T
    v8f acc = {0,0,0,0,0,0,0,0};
    for (int kk = 0; kk < 8; ++kk)
      acc = wmma_bf16(load_a_frag(z_lds, D_, kk * 32, lane),
                      load_b_frag(wkey, D_, c0, kk * 32, lane), acc);
#pragma unroll
    for (int j = 0; j < 8; ++j) keys[(mb + j) * D_ + cl] = acc[j];
  }
  { // values = z @ W_value^T
    v8f acc = {0,0,0,0,0,0,0,0};
    for (int kk = 0; kk < 8; ++kk)
      acc = wmma_bf16(load_a_frag(z_lds, D_, kk * 32, lane),
                      load_b_frag(wval, D_, c0, kk * 32, lane), acc);
#pragma unroll
    for (int j = 0; j < 8; ++j) vals[(mb + j) * D_ + cl] = acc[j];
  }
  __syncthreads();

  for (int v = 0; v < NVAR; ++v) {
    { // queries_v = z @ W_query[v]^T + b_query[v]
      const __bf16* wqv = wq + (size_t)v * D_ * D_;
      v8f acc = {0,0,0,0,0,0,0,0};
      for (int kk = 0; kk < 8; ++kk)
        acc = wmma_bf16(load_a_frag(z_lds, D_, kk * 32, lane),
                        load_b_frag(wqv, D_, c0, kk * 32, lane), acc);
      const float bias = bq[v * D_ + cl];
#pragma unroll
      for (int j = 0; j < 8; ++j) q[(mb + j) * D_ + cl] = acc[j] + bias;
    }
    __syncthreads();

    { // partial dot(q_row, keys_row)
      float p = 0.f;
      const int cb = seg * 8;
#pragma unroll
      for (int i = 0; i < 8; ++i) p += q[row * D_ + cb + i] * keys[row * D_ + cb + i];
      red[row * 32 + seg] = p;
    }
    __syncthreads();
    if (tid < 16) {
      float s = 0.f;
      for (int i = 0; i < 32; ++i) s += red[tid * 32 + i];
      attn_s[tid] = 1.0f / (1.0f + __expf(-s * (1.0f / 16.0f)));  // scale = sqrt(256)
    }
    __syncthreads();

    { // slot mix + sum of squares
      const float a = attn_s[row];
      const int cb = seg * 8;
      float ss = 0.f;
#pragma unroll
      for (int i = 0; i < 8; ++i) {
        float s = a * vals[row * D_ + cb + i] + (1.0f - a) * q[row * D_ + cb + i];
        q[row * D_ + cb + i] = s;
        ss += s * s;
      }
      red[row * 32 + seg] = ss;
    }
    __syncthreads();
    if (tid < 16) {
      float s = 0.f;
      for (int i = 0; i < 32; ++i) s += red[tid * 32 + i];
      inv_s[tid] = 1.0f / fmaxf(sqrtf(s), 1e-8f);
    }
    __syncthreads();

    { // normalized base slots -> out[(r*16+v)*256 + d]
      const float inv = inv_s[row];
      const int cb = seg * 8;
      float* dst = out + ((size_t)(r0 + row) * NVAR + v) * D_ + cb;
#pragma unroll
      for (int i = 0; i < 8; ++i) dst[i] = q[row * D_ + cb + i] * inv;
    }
    __syncthreads();
  }
}

// ---- Kernel 3: sequential temporal scan (rows independent) --------------
// grid = B workgroups; each handles the 16 slot-vars of one batch element.
// W_temporal^T kept in registers per wave (8x v16bf). State ping-pongs in LDS.

__global__ __launch_bounds__(512) void cjepa_scan(
    float* __restrict__ out, const __bf16* __restrict__ wtmp) {
  __shared__ __align__(16) __bf16 prev[2][16 * D_];   // 2 x 8 KB
  const int tid  = threadIdx.x;
  const int b    = blockIdx.x;
  const int lane = tid & 31;
  const int c0   = (tid >> 5) * 16;
  const int cl   = c0 + (lane & 15);
  const int mb   = (lane >> 4) << 3;

  // Register-resident B fragments of W_temporal^T for this wave's column tile
  v16bf bfrag[8];
#pragma unroll
  for (int kk = 0; kk < 8; ++kk)
    bfrag[kk] = load_b_frag(wtmp, D_, c0, kk * 32, lane);

  { // load slots at t=0 (written by cjepa_slots) into prev[0]
    const float* s0 = out + (size_t)b * T_ * NVAR * D_;
    for (int i = 0; i < 8; ++i) {
      int e = tid + i * 512;
      prev[0][e] = (__bf16)s0[e];
    }
  }
  __syncthreads();

  for (int t = 1; t < T_; ++t) {
    const __bf16* src = prev[(t - 1) & 1];
    __bf16*       dst = prev[t & 1];

    v8f acc = {0,0,0,0,0,0,0,0};
#pragma unroll
    for (int kk = 0; kk < 8; ++kk)
      acc = wmma_bf16(load_a_frag(src, D_, kk * 32, lane), bfrag[kk], acc);

    float* orow = out + ((size_t)(b * T_ + t) * NVAR) * D_;
#pragma unroll
    for (int j = 0; j < 8; ++j) {
      const int m = mb + j;                       // slot-var index n
      float tmp = tanhf(acc[j]);
      size_t idx = (size_t)m * D_ + cl;
      float nv = ALPHA_ * orow[idx] + (1.0f - ALPHA_) * tmp;
      orow[idx] = nv;
      dst[idx] = (__bf16)nv;
    }
    __syncthreads();
  }
}

// ---- Launch -------------------------------------------------------------

extern "C" void kernel_launch(void* const* d_in, const int* in_sizes, int n_in,
                              void* d_out, int out_size, void* d_ws, size_t ws_size,
                              hipStream_t stream) {
  const float* obs  = (const float*)d_in[0];  // (B,T,DOBS)
  const float* Wenc = (const float*)d_in[1];  // (D,DOBS)
  const float* benc = (const float*)d_in[2];  // (D)
  const float* Wkey = (const float*)d_in[3];  // (D,D)
  const float* Wval = (const float*)d_in[4];  // (D,D)
  const float* Wq   = (const float*)d_in[5];  // (N,D,D)
  const float* bq   = (const float*)d_in[6];  // (N,D)
  const float* Wtmp = (const float*)d_in[7];  // (D,D)
  float* out = (float*)d_out;                 // (B,T,N,D)

  char* ws = (char*)d_ws;
  __bf16* wencb = (__bf16*)(ws + 0);          //  262144 elems
  __bf16* wkeyb = (__bf16*)(ws + 524288);     //   65536
  __bf16* wvalb = (__bf16*)(ws + 655360);     //   65536
  __bf16* wqb   = (__bf16*)(ws + 786432);     // 1048576
  __bf16* wtmpb = (__bf16*)(ws + 2883584);    //   65536
  __bf16* zb    = (__bf16*)(ws + 3014656);    // 2097152 (B*T*D)

  cjepa_cvt_bf16<<<(D_*DOBS + 255) / 256, 256, 0, stream>>>(Wenc, wencb, D_*DOBS);
  cjepa_cvt_bf16<<<(D_*D_   + 255) / 256, 256, 0, stream>>>(Wkey, wkeyb, D_*D_);
  cjepa_cvt_bf16<<<(D_*D_   + 255) / 256, 256, 0, stream>>>(Wval, wvalb, D_*D_);
  cjepa_cvt_bf16<<<(NVAR*D_*D_ + 255) / 256, 256, 0, stream>>>(Wq, wqb, NVAR*D_*D_);
  cjepa_cvt_bf16<<<(D_*D_   + 255) / 256, 256, 0, stream>>>(Wtmp, wtmpb, D_*D_);

  const int rowTiles = (B_ * T_) / 16;  // 512
  cjepa_encode<<<rowTiles, 512, 0, stream>>>(obs, wencb, benc, zb);
  cjepa_slots <<<rowTiles, 512, 0, stream>>>(zb, wkeyb, wvalb, wqb, bq, out);
  cjepa_scan  <<<B_,       512, 0, stream>>>(out, wtmpb);
}
